// MyModel_87522843559391
// MI455X (gfx1250) — matrix operand order repaired and checked by reading, not compiled, not run on hardware
//
#include <hip/hip_runtime.h>
#include <cstdint>

typedef __attribute__((ext_vector_type(16))) _Float16 v16h;
typedef __attribute__((ext_vector_type(8)))  float    v8f;
typedef __attribute__((ext_vector_type(8)))  uint32_t v8u;

#define BATCH 4096
#define SEQ   512
#define EMBD  10
#define HID   32

__device__ __forceinline__ float fast_tanh(float x) {
#if __has_builtin(__builtin_amdgcn_tanhf)
  return __builtin_amdgcn_tanhf(x);
#elif __has_builtin(__builtin_amdgcn_tanh_f32)
  return __builtin_amdgcn_tanh_f32(x);
#else
  return tanhf(x);
#endif
}

// A_x (16x32 f16) from an embedding row, per ISA A-layout lane-half K split:
// half0 lanes: K=0..7 live (emb[0..7]); half1 lanes: K=8,9 live (emb[8..9]),
// plus K=10 (element ax[2]) carries the constant 1.0 that multiplies the bias
// row folded into B_wx. Everything else zero.
__device__ __forceinline__ v16h build_ax(bool hh, float2 e0, float2 e1, float2 e2,
                                         float2 e3, float2 e4) {
  v16h ax = {};
  ax[0] = (_Float16)(hh ? e4.x : e0.x);
  ax[1] = (_Float16)(hh ? e4.y : e0.y);
  ax[2] = (_Float16)(hh ? 1.0f : e1.x);   // K=10 (half1) = 1.0 -> bias row
  ax[3] = (_Float16)(hh ? 0.0f : e1.y);
  ax[4] = (_Float16)(hh ? 0.0f : e2.x);
  ax[5] = (_Float16)(hh ? 0.0f : e2.y);
  ax[6] = (_Float16)(hh ? 0.0f : e3.x);
  ax[7] = (_Float16)(hh ? 0.0f : e3.y);
  return ax;
}

// One wave = 16 batch rows, full 512-step scan. 8 waves/block, 32 blocks = 256 waves.
__global__ __launch_bounds__(256) void rnn_scan_fused(
    const int* __restrict__ tokens,   // (4096, 512)
    const float* __restrict__ emb,    // (10000, 10)
    const float* __restrict__ Wx,     // (10, 32)
    const float* __restrict__ Wh,     // (32, 32)
    const float* __restrict__ bias,   // (32,)
    const float* __restrict__ Wd,     // (32,)
    const float* __restrict__ bd,     // (1,)
    float* __restrict__ out)          // (4096,)
{
  __shared__ __align__(16) _Float16 hbuf[8][512];   // per-wave 16x32 f16 h tile (1KB each)

  const int tid  = threadIdx.x;
  const int lane = tid & 31;
  const int wave = tid >> 5;
  const int half = lane >> 4;
  const int l16  = lane & 15;
  const bool hh  = (half != 0);

  const int tile   = blockIdx.x * 8 + wave;
  const int batch0 = tile * 16;

  // ---- constant WMMA operands, VGPR-resident for the whole scan ----
  // B_wh tiles (32x16 f16): N = hid_out = l16 (+16 for tile 1), element e = K = hid_in in table order
  v16h bwh0, bwh1;
  {
#pragma unroll
    for (int e = 0; e < 16; ++e) {
      const int k = (e < 8) ? 8 * half + e : 16 + 8 * half + (e - 8);
      bwh0[e] = (_Float16)Wh[k * HID + l16];
      bwh1[e] = (_Float16)Wh[k * HID + 16 + l16];
    }
  }
  // B_wx tiles: N = l16 (+16), element e = K = emb dim in table order (0..9 live). Row K=10 carries the bias
  // (multiplied by the constant 1.0 planted at A's K=10), rest zero-padded.
  v16h bwx0 = {}, bwx1 = {};
#pragma unroll
  for (int e = 0; e < 16; ++e) {
    const int k = (e < 8) ? 8 * half + e : 16 + 8 * half + (e - 8);
    if (k < EMBD) {
      bwx0[e] = (_Float16)Wx[k * HID + l16];
      bwx1[e] = (_Float16)Wx[k * HID + 16 + l16];
    } else if (k == EMBD) {
      bwx0[e] = (_Float16)bias[l16];
      bwx1[e] = (_Float16)bias[16 + l16];
    }
  }
  const v8f cz = {};   // xproj C operand: zero (inline SRC2=0)

  // LDS addressing for the per-step h repack (same-wave, DS ops are in-order)
  _Float16* hw = &hbuf[wave][0] + half * 256 + l16;                              // D-layout store base
  const uint32_t* hr = (const uint32_t*)(&hbuf[wave][0] + l16 * 32 + 8 * half);  // A-layout reload base

  // ---- pipeline prologue: dx ready for t=0, emb row for t=1, token for t=2 ----
  const int* tokp = tokens + (size_t)(batch0 + l16) * SEQ;  // lane halves duplicate (same line)
  float2 ep0, ep1, ep2, ep3, ep4;   // emb row for step t+1
  v8f dxc0, dxc1;                   // xproj tiles for step t (bias included)
  int tok_n;                        // token for step t+2
  {
    int tk = tokp[0];
    const float2* er = (const float2*)(emb + (size_t)tk * EMBD);
    float2 q0 = er[0], q1 = er[1], q2 = er[2], q3 = er[3], q4 = er[4];
    v16h ax0 = build_ax(hh, q0, q1, q2, q3, q4);
    dxc0 = __builtin_amdgcn_wmma_f32_16x16x32_f16(false, ax0, false, bwx0, (short)0, cz, false, false);
    dxc1 = __builtin_amdgcn_wmma_f32_16x16x32_f16(false, ax0, false, bwx1, (short)0, cz, false, false);

    tk = tokp[1];
    const float2* er1 = (const float2*)(emb + (size_t)tk * EMBD);
    ep0 = er1[0]; ep1 = er1[1]; ep2 = er1[2]; ep3 = er1[3]; ep4 = er1[4];

    tok_n = tokp[2];
  }

  v16h ah = {};   // h0 = 0

  // Unroll-2: ping-pongs the pipeline registers across the two copies (no rotation
  // movs) and separates each xproj WMMA from its consumer by a full tanh/DS block.
#pragma unroll 2
  for (int t = 0; t < SEQ; ++t) {
    // ---- critical chain: recurrence WMMAs consume pre-computed xproj tiles ----
    v8f d0 = __builtin_amdgcn_wmma_f32_16x16x32_f16(false, ah, false, bwh0, (short)0, dxc0, false, false);
    v8f d1 = __builtin_amdgcn_wmma_f32_16x16x32_f16(false, ah, false, bwh1, (short)0, dxc1, false, false);

    // ---- off-chain: prefetch token(t+3), gather emb(t+2), xproj WMMAs for t+1 ----
    const int t3 = (t + 3 < SEQ) ? t + 3 : SEQ - 1;
    const int tok_n2 = tokp[t3];
    const float2* ern = (const float2*)(emb + (size_t)tok_n * EMBD);
    const float2 n0 = ern[0], n1 = ern[1], n2 = ern[2], n3 = ern[3], n4 = ern[4];

    v16h ax = build_ax(hh, ep0, ep1, ep2, ep3, ep4);
    v8f dxn0 = __builtin_amdgcn_wmma_f32_16x16x32_f16(false, ax, false, bwx0, (short)0, cz, false, false);
    v8f dxn1 = __builtin_amdgcn_wmma_f32_16x16x32_f16(false, ax, false, bwx1, (short)0, cz, false, false);

    // ---- h = tanh(d); f16 store to [batch][hid] (slot: batch = r+8*half, hid = l16 (+16)) ----
#pragma unroll
    for (int r = 0; r < 8; ++r) {
      hw[r * 32]      = (_Float16)fast_tanh(d0[r]);
      hw[r * 32 + 16] = (_Float16)fast_tanh(d1[r]);
    }
    asm volatile("s_wait_dscnt 0" ::: "memory");

    // reload h in A-matrix 16x32 layout: dwords {0..3} and {8..11} of the lane's K row
    v8u hv;
    hv[0] = hr[0]; hv[1] = hr[1]; hv[2] = hr[2];  hv[3] = hr[3];
    hv[4] = hr[8]; hv[5] = hr[9]; hv[6] = hr[10]; hv[7] = hr[11];
    ah = __builtin_bit_cast(v16h, hv);

    // ---- rotate pipeline (renamed away by the unroll-2 copies) ----
    dxc0 = dxn0; dxc1 = dxn1;
    ep0 = n0; ep1 = n1; ep2 = n2; ep3 = n3; ep4 = n4;
    tok_n = tok_n2;
  }

  // ---- head: out[b] = sigmoid(h_last . Wd + bd); h_last read back from LDS ----
  asm volatile("s_wait_dscnt 0" ::: "memory");
  if (lane < 16) {
    const uint32_t* hrow = (const uint32_t*)(&hbuf[wave][0] + lane * 32);
    float s = bd[0];
#pragma unroll
    for (int k = 0; k < 16; ++k) {
      const uint32_t w = hrow[k];
      const float lo = (float)__builtin_bit_cast(_Float16, (unsigned short)(w & 0xFFFFu));
      const float hi = (float)__builtin_bit_cast(_Float16, (unsigned short)(w >> 16));
      s += lo * Wd[2 * k] + hi * Wd[2 * k + 1];   // Wd uniform -> scalar loads
    }
    out[batch0 + lane] = 1.0f / (1.0f + __expf(-s));
  }
}

extern "C" void kernel_launch(void* const* d_in, const int* in_sizes, int n_in,
                              void* d_out, int out_size, void* d_ws, size_t ws_size,
                              hipStream_t stream) {
  (void)in_sizes; (void)n_in; (void)out_size; (void)d_ws; (void)ws_size;
  const int*   tokens = (const int*)  d_in[0];
  const float* emb    = (const float*)d_in[1];
  const float* Wx     = (const float*)d_in[2];
  const float* Wh     = (const float*)d_in[3];
  const float* b      = (const float*)d_in[4];
  const float* Wd     = (const float*)d_in[5];
  const float* bd     = (const float*)d_in[6];
  float* out = (float*)d_out;

  dim3 grid(BATCH / (16 * 8));   // 32 blocks
  dim3 block(256);               // 8 waves of 16 batch rows each
  rnn_scan_fused<<<grid, block, 0, stream>>>(tokens, emb, Wx, Wh, b, Wd, bd, out);
}
